// SoftTripletLoss_2705829397240
// MI455X (gfx1250) — matrix operand order repaired and checked by reading, not compiled, not run on hardware
//
#include <hip/hip_runtime.h>
#include <stdint.h>

typedef __attribute__((ext_vector_type(16))) __bf16 v16bf;
typedef __attribute__((ext_vector_type(8)))  float  v8f;

#define N_ROWS 4096
#define DIMK   2048
#define MB     32      // rows per block
#define NB     128     // cols per n-block tile
#define KC     64      // k chunk staged in LDS
#define NKC    (DIMK / KC)
#define LDP    72      // padded LDS row stride (bf16 elements) -> conflict-free b128

union FragBF { v16bf v; uint4 q[2]; };
union AccF   { v8f  v; float f[8]; };

__device__ __forceinline__ unsigned short f32_to_bf16(float f) {
    unsigned u = __float_as_uint(f);
    u += 0x7FFFu + ((u >> 16) & 1u);   // round-to-nearest-even
    return (unsigned short)(u >> 16);
}

// CDNA5 async global->LDS copy (no VGPR round trip, tracked by ASYNCcnt).
// VDST = LDS byte offset (low 32 bits of the flat-cast shared pointer).
__device__ __forceinline__ void async_copy_b128(const unsigned short* lds_dst,
                                                const unsigned short* gsrc) {
    unsigned lds_off = (unsigned)(uintptr_t)lds_dst;
    unsigned long long ga = (unsigned long long)(uintptr_t)gsrc;
    asm volatile("global_load_async_to_lds_b128 %0, %1, off"
                 :: "v"(lds_off), "v"(ga) : "memory");
}
__device__ __forceinline__ void wait_asynccnt0() {
    asm volatile("s_wait_asynccnt 0x0" ::: "memory");
}

// ---------------- Kernel 1: L2-normalize rows, emit bf16 ----------------
__global__ __launch_bounds__(256) void norm_rows_kernel(
        const float* __restrict__ X, unsigned short* __restrict__ Xn) {
    __shared__ float red[256];
    const int row = blockIdx.x;
    const int t = threadIdx.x;
    const float* xr = X + (size_t)row * DIMK;
    float v[8];
    float ss = 0.f;
#pragma unroll
    for (int i = 0; i < 8; ++i) {
        v[i] = xr[t + i * 256];
        ss += v[i] * v[i];
    }
    red[t] = ss;
    __syncthreads();
    for (int s = 128; s > 0; s >>= 1) {
        if (t < s) red[t] += red[t + s];
        __syncthreads();
    }
    const float inv = 1.0f / (sqrtf(red[0]) + 1e-12f);
    unsigned short* outr = Xn + (size_t)row * DIMK;
#pragma unroll
    for (int i = 0; i < 8; ++i)
        outr[t + i * 256] = f32_to_bf16(v[i] * inv);
}

// ------- Kernel 2: fused Gram (bf16 WMMA) + masked soft pos/neg pooling -------
__global__ __launch_bounds__(256) void gram_softloss_kernel(
        const unsigned short* __restrict__ Xn,
        const int* __restrict__ targets,
        float* __restrict__ rowloss) {
    __shared__ unsigned short ldsA[2][MB * LDP];   // 2 x  4,608 B
    __shared__ unsigned short ldsB[2][NB * LDP];   // 2 x 18,432 B
    __shared__ float4 Pbuf[8][MB];                 //      4,096 B

    const int t    = threadIdx.x;
    const int wv   = t >> 5;
    const int lane = t & 31;
    const int half = lane >> 4;   // 0: lanes 0-15, 1: lanes 16-31
    const int lrow = lane & 15;
    const int mi   = wv & 1;      // 16-row group within block
    const int ni2  = wv >> 1;     // 32-col strip within 128-col tile
    const int Rb   = blockIdx.x * MB;

    // per-thread staging coordinates (constant across all stages)
    const int sArow = t >> 3;              // 0..31
    const int sAkk  = (t & 7) * 8;         // 0..56
    // zero cross-wave partial buffer (8 waves * 32 rows = 256 slots)
    Pbuf[wv][lane] = make_float4(0.f, 0.f, 0.f, 0.f);

    // row targets for this lane's 8 C rows (C layout: VGPR g -> M = g + 8*half)
    const int rbase = Rb + mi * 16 + half * 8;
    int trow[8];
#pragma unroll
    for (int g = 0; g < 8; ++g) trow[g] = targets[rbase + g];

    float pn[8], pd[8], nn[8], nd[8];
#pragma unroll
    for (int g = 0; g < 8; ++g) { pn[g] = pd[g] = nn[g] = nd[g] = 0.f; }

    for (int nb = 0; nb < N_ROWS; nb += NB) {
        AccF acc0, acc1;
#pragma unroll
        for (int g = 0; g < 8; ++g) { acc0.f[g] = 0.f; acc1.f[g] = 0.f; }

        // prologue: async-stage k-chunk 0 into buffer 0
        async_copy_b128(&ldsA[0][sArow * LDP + sAkk],
                        Xn + (size_t)(Rb + sArow) * DIMK + 0 + sAkk);
#pragma unroll
        for (int i = 0; i < 4; ++i) {
            const int q = i * 256 + t;
            const int row = q >> 3;
            const int kk  = (q & 7) * 8;
            async_copy_b128(&ldsB[0][row * LDP + kk],
                            Xn + (size_t)(nb + row) * DIMK + 0 + kk);
        }

        for (int ik = 0; ik < NKC; ++ik) {
            const int buf = ik & 1;
            wait_asynccnt0();     // stage ik landed in LDS[buf]
            __syncthreads();      // all waves staged; prev buffer reads retired

            if (ik + 1 < NKC) {   // async-stage ik+1 into the other buffer
                const int nbuf = buf ^ 1;
                const int kb = (ik + 1) * KC;
                async_copy_b128(&ldsA[nbuf][sArow * LDP + sAkk],
                                Xn + (size_t)(Rb + sArow) * DIMK + kb + sAkk);
#pragma unroll
                for (int i = 0; i < 4; ++i) {
                    const int q = i * 256 + t;
                    const int row = q >> 3;
                    const int kk  = (q & 7) * 8;
                    async_copy_b128(&ldsB[nbuf][row * LDP + kk],
                                    Xn + (size_t)(nb + row) * DIMK + kb + kk);
                }
            }

#pragma unroll
            for (int ks = 0; ks < 2; ++ks) {
                // A fragment 16x32 bf16: lane<16 -> K 0..15, lane>=16 -> K 16..31
                FragBF a;
                const int arow = mi * 16 + lrow;
                const int ka   = ks * 32 + half * 16;
                a.q[0] = *(const uint4*)&ldsA[buf][arow * LDP + ka];
                a.q[1] = *(const uint4*)&ldsA[buf][arow * LDP + ka + 8];
                // B fragment 32x16 bf16: lane<16 -> K 0..15 of column lane,
                // lane>=16 -> K 16..31 of column lane-16 (B[k][n] = Xn[n][k])
                FragBF b0, b1;
                const int kbl   = ks * 32 + half * 16;
                const int brow0 = ni2 * 32 + lrow;
                b0.q[0] = *(const uint4*)&ldsB[buf][brow0 * LDP + kbl];
                b0.q[1] = *(const uint4*)&ldsB[buf][brow0 * LDP + kbl + 8];
                const int brow1 = brow0 + 16;
                b1.q[0] = *(const uint4*)&ldsB[buf][brow1 * LDP + kbl];
                b1.q[1] = *(const uint4*)&ldsB[buf][brow1 * LDP + kbl + 8];

                acc0.v = __builtin_amdgcn_wmma_f32_16x16x32_bf16(
                    false, a.v, false, b0.v, (short)0, acc0.v, false, false);
                acc1.v = __builtin_amdgcn_wmma_f32_16x16x32_bf16(
                    false, a.v, false, b1.v, (short)0, acc1.v, false, false);
            }
        }

        // epilogue: dist = sqrt(max(2 - 2*gram, eps)); fold into soft pos/neg
        const int col0 = nb + ni2 * 32 + lrow;
        const int col1 = col0 + 16;
        const int tc0 = targets[col0];
        const int tc1 = targets[col1];
#pragma unroll
        for (int g = 0; g < 8; ++g) {
            const int grow = rbase + g;
            {
                const float dist = sqrtf(fmaxf(2.0f - 2.0f * acc0.f[g], 1e-12f));
                const bool same = (trow[g] == tc0);
                if (same && (grow != col0)) {
                    const float e = __expf(dist);  pn[g] += e * dist; pd[g] += e;
                }
                if (!same) {
                    const float e = __expf(-dist); nn[g] += e * dist; nd[g] += e;
                }
            }
            {
                const float dist = sqrtf(fmaxf(2.0f - 2.0f * acc1.f[g], 1e-12f));
                const bool same = (trow[g] == tc1);
                if (same && (grow != col1)) {
                    const float e = __expf(dist);  pn[g] += e * dist; pd[g] += e;
                }
                if (!same) {
                    const float e = __expf(-dist); nn[g] += e * dist; nd[g] += e;
                }
            }
        }
    }

    // reduce over the 16 lanes holding one row's columns (xor stays in half)
#pragma unroll
    for (int g = 0; g < 8; ++g) {
#pragma unroll
        for (int m = 1; m <= 8; m <<= 1) {
            pn[g] += __shfl_xor(pn[g], m, 32);
            pd[g] += __shfl_xor(pd[g], m, 32);
            nn[g] += __shfl_xor(nn[g], m, 32);
            nd[g] += __shfl_xor(nd[g], m, 32);
        }
    }
    if (lrow == 0) {  // lanes 0 and 16 carry the half-group sums
#pragma unroll
        for (int g = 0; g < 8; ++g) {
            const int rl = mi * 16 + half * 8 + g;
            Pbuf[wv][rl] = make_float4(pn[g], pd[g], nn[g], nd[g]);
        }
    }
    __syncthreads();

    if (t < MB) {  // deterministic cross-wave combine, one thread per row
        float4 s = make_float4(0.f, 0.f, 0.f, 0.f);
#pragma unroll
        for (int w = 0; w < 8; ++w) {
            const float4 p = Pbuf[w][t];
            s.x += p.x; s.y += p.y; s.z += p.z; s.w += p.w;
        }
        const float pos = (s.y > 0.f) ? s.x / s.y : 0.f;
        const float neg = (s.w > 0.f) ? s.z / s.w : 0.f;
        rowloss[Rb + t] = fmaxf(pos + 0.3f - neg, 0.f);
    }
}

// ---------------- Kernel 3: deterministic mean over 4096 row losses ----------------
__global__ __launch_bounds__(256) void reduce_mean_kernel(
        const float* __restrict__ rowloss, float* __restrict__ out) {
    __shared__ float red[256];
    const int t = threadIdx.x;
    float s = 0.f;
    for (int i = t; i < N_ROWS; i += 256) s += rowloss[i];
    red[t] = s;
    __syncthreads();
    for (int k = 128; k > 0; k >>= 1) {
        if (t < k) red[t] += red[t + k];
        __syncthreads();
    }
    if (t == 0) out[0] = red[0] / (float)N_ROWS;
}

extern "C" void kernel_launch(void* const* d_in, const int* in_sizes, int n_in,
                              void* d_out, int out_size, void* d_ws, size_t ws_size,
                              hipStream_t stream) {
    (void)in_sizes; (void)n_in; (void)out_size; (void)ws_size;
    const float* X       = (const float*)d_in[0];
    const int*   targets = (const int*)d_in[1];

    unsigned short* Xn = (unsigned short*)d_ws;                       // 16.78 MB bf16
    float* rowloss = (float*)((char*)d_ws +
                     (size_t)N_ROWS * DIMK * sizeof(unsigned short)); // 16 KB
    float* out = (float*)d_out;

    norm_rows_kernel<<<N_ROWS, 256, 0, stream>>>(X, Xn);
    gram_softloss_kernel<<<N_ROWS / MB, 256, 0, stream>>>(Xn, targets, rowloss);
    reduce_mean_kernel<<<1, 256, 0, stream>>>(rowloss, out);
}
